// CAConv2_18769007084594
// MI455X (gfx1250) — matrix operand order repaired and checked by reading, not compiled, not run on hardware
//
#include <hip/hip_runtime.h>
#include <hip/hip_bf16.h>

typedef __attribute__((ext_vector_type(16))) __bf16 v16bf;
typedef __attribute__((ext_vector_type(8)))  float  v8f;

#define EPSV 1e-5f

__device__ __forceinline__ unsigned short f2bf(float f){
    unsigned int u = __float_as_uint(f);
    u += 0x7FFFu + ((u >> 16) & 1u);          // round-to-nearest-even
    return (unsigned short)(u >> 16);
}

// ---------------- Stage 1: directional pooling ----------------
// grid: N*C = 1024 blocks, block: 128 threads (4 waves)
// y_buf[(n*128+c)*256 + l] : l<128 -> x_h[h], l>=128 -> x_w[w]
__global__ __launch_bounds__(128) void pool_kernel(const float* __restrict__ x,
                                                   float* __restrict__ y_buf){
    const int nc  = blockIdx.x;
    const float* p = x + (size_t)nc * (128 * 128);
    const int tid = threadIdx.x;
    // Phase A: column sums (over h) -> x_w[w], thread = w (coalesced)
    {
        float s1 = 0.f, s2 = 0.f, s3 = 0.f;
        for (int h = 0; h < 128; ++h){
            float v = p[h * 128 + tid];
            s1 += v;
            if (h < 64) s2 += v;
            if (h < 32) s3 += v;
        }
        y_buf[nc * 256 + 128 + tid] = s1 * (1.f/128.f) + s2 * (1.f/64.f) + s3 * (1.f/32.f);
    }
    // Phase B: row sums (over w) -> x_h[h], one wave32 per row, shfl reduce
    const int wave = tid >> 5, lane = tid & 31;
    for (int k = 0; k < 32; ++k){
        int h = wave + k * 4;
        float v0 = p[h * 128 + lane];
        float v1 = p[h * 128 + lane + 32];
        float v2 = p[h * 128 + lane + 64];
        float v3 = p[h * 128 + lane + 96];
        float s1 = v0 + v1 + v2 + v3;   // w in [0,128)
        float s2 = v0 + v1;             // w in [0,64)
        float s3 = v0;                  // w in [0,32)
        for (int off = 16; off > 0; off >>= 1){
            s1 += __shfl_xor(s1, off, 32);
            s2 += __shfl_xor(s2, off, 32);
            s3 += __shfl_xor(s3, off, 32);
        }
        if (lane == 0)
            y_buf[nc * 256 + h] = s1 * (1.f/128.f) + s2 * (1.f/64.f) + s3 * (1.f/32.f);
    }
}

// ---------------- Stage 2: squeeze MLP + sigmoid attention ----------------
// grid: N = 8 blocks, block: 256 threads
__global__ __launch_bounds__(256) void attn_kernel(
    const float* __restrict__ y_buf,
    const float* __restrict__ w1, const float* __restrict__ b1,
    const float* __restrict__ g1, const float* __restrict__ be1,
    const float* __restrict__ m1, const float* __restrict__ v1,
    const float* __restrict__ wh, const float* __restrict__ bh,
    const float* __restrict__ ww, const float* __restrict__ bw,
    float* __restrict__ a_buf){
    __shared__ float yh[8 * 256];
    const int n = blockIdx.x;
    const int l = threadIdx.x;
    float acc[8];
    #pragma unroll
    for (int m = 0; m < 8; ++m) acc[m] = b1[m];
    for (int c = 0; c < 128; ++c){
        float yv = y_buf[(n * 128 + c) * 256 + l];
        #pragma unroll
        for (int m = 0; m < 8; ++m) acc[m] += w1[m * 128 + c] * yv;
    }
    #pragma unroll
    for (int m = 0; m < 8; ++m){
        float s  = g1[m] * rsqrtf(v1[m] + EPSV);
        float t  = acc[m] * s + (be1[m] - m1[m] * s);
        float hs = t * fminf(fmaxf(t + 3.f, 0.f), 6.f) * (1.f / 6.f); // h_swish
        yh[m * 256 + l] = hs;
    }
    __syncthreads();
    const bool  isH  = (l < 128);          // wave-uniform (waves 0-3 vs 4-7)
    const float* wsel = isH ? wh : ww;
    const float* bsel = isH ? bh : bw;
    for (int c = 0; c < 128; ++c){
        float a = bsel[c];
        #pragma unroll
        for (int m = 0; m < 8; ++m) a += wsel[c * 8 + m] * yh[m * 256 + l];
        a = 1.f / (1.f + __expf(-a));      // sigmoid
        a_buf[(n * 128 + c) * 256 + l] = a;
    }
}

// ---------------- Weight / scale prep ----------------
// WB layout: [t(9)][kc(4)][o(128)][c32(32)] bf16  (fragment-ready)
__global__ __launch_bounds__(256) void prep_w_kernel(const float* __restrict__ wc,
                                                     unsigned short* __restrict__ WB){
    int i = blockIdx.x * 256 + threadIdx.x;
    if (i >= 9 * 4 * 128 * 32) return;
    int c32 = i & 31;
    int o   = (i >> 5) & 127;
    int kc  = (i >> 12) & 3;
    int t   = i >> 14;
    int kh = t / 3, kw = t % 3;
    int c  = kc * 32 + c32;
    WB[i] = f2bf(wc[((o * 128 + c) * 3 + kh) * 3 + kw]);
}

// s2t2[o] = scale, s2t2[128+o] = bias  (folds conv bias + BN)
__global__ void prep_s_kernel(const float* __restrict__ bc,
                              const float* __restrict__ g2, const float* __restrict__ be2,
                              const float* __restrict__ m2, const float* __restrict__ v2,
                              float* __restrict__ s2t2){
    int o = threadIdx.x;
    float s = g2[o] * rsqrtf(v2[o] + EPSV);
    s2t2[o]       = s;
    s2t2[128 + o] = bc[o] * s + be2[o] - m2[o] * s;
}

// ---------------- Stage 3: modulate + NCHW->NHWC bf16 ----------------
// grid: N*H*4 = 4096 blocks, block 256. xmod[n][h][w][c] = bf16(x*a_h*a_w)
__global__ __launch_bounds__(256) void mod_kernel(const float* __restrict__ x,
                                                  const float* __restrict__ a_buf,
                                                  unsigned int* __restrict__ xmod_u32){
    __shared__ float tile[128 * 33];       // [c][j], padded
    int b  = blockIdx.x;
    int wt = b & 3, h = (b >> 2) & 127, n = b >> 9;
    int w0 = wt * 32;
    int tid = threadIdx.x;
    #pragma unroll
    for (int k = 0; k < 16; ++k){
        int e = k * 256 + tid;
        int c = e >> 5, j = e & 31;
        float v  = x[((n * 128 + c) * 128 + h) * 128 + w0 + j];
        float ah = a_buf[(n * 128 + c) * 256 + h];
        float aw = a_buf[(n * 128 + c) * 256 + 128 + w0 + j];
        tile[c * 33 + j] = v * ah * aw;
    }
    __syncthreads();
    #pragma unroll
    for (int k = 0; k < 8; ++k){
        int e = k * 256 + tid;
        int j = e >> 6, cp = e & 63;       // cp = channel pair
        float v0 = tile[(2 * cp    ) * 33 + j];
        float v1 = tile[(2 * cp + 1) * 33 + j];
        unsigned int u = (unsigned int)f2bf(v0) | ((unsigned int)f2bf(v1) << 16);
        xmod_u32[(((n * 128 + h) * 128 + w0 + j)) * 64 + cp] = u;   // coalesced in cp
    }
}

// ---------------- Stage 4: implicit-GEMM 3x3 conv via WMMA bf16 ----------------
union FragU { uint4 q[2]; v16bf v; };

// grid: N*H = 1024 blocks, block 256 (8 waves; wave = 16-pixel tile)
// Software-pipelined: global weight slice prefetched 1 step ahead into regs,
// double-buffered LDS, 1 barrier/K-step, DS clause fenced ahead of WMMA chain.
__global__ __launch_bounds__(256) void conv_kernel(const unsigned short* __restrict__ xmod,
                                                   const unsigned short* __restrict__ WB,
                                                   const float* __restrict__ s2t2,
                                                   float* __restrict__ out){
    __shared__ __align__(16) unsigned short sWB[2][128 * 40]; // [o][c'] rows padded to 40
    const int nh   = blockIdx.x;
    const int n    = nh >> 7, h = nh & 127;
    const int tid  = threadIdx.x;
    const int wave = tid >> 5, lane = tid & 31;
    const int hl   = lane >> 4, lm = lane & 15;
    const int wpix = wave * 16 + lm;       // this lane's output pixel column

    v8f acc[8];
    #pragma unroll
    for (int i = 0; i < 8; ++i)
        #pragma unroll
        for (int j = 0; j < 8; ++j) acc[i][j] = 0.f;

    // per-thread staging registers (2 x uint4 = one 8 KiB slice / 256 threads)
    uint4 gc0, gc1;                        // current slice data
    {
        const uint4* src = (const uint4*)WB;
        gc0 = src[tid];
        gc1 = src[256 + tid];
    }
    // LDS destination (padded rows, 16B aligned: row stride 20 words)
    const int eo0 = tid,        ec0 = tid & 3,        er0 = tid >> 2;
    const int er1 = (256 + tid) >> 2, ec1 = (256 + tid) & 3;
    (void)eo0;

    for (int idx = 0; idx < 36; ++idx){    // idx = t*4 + kc over 9 taps x 4 chunks
        const int buf = idx & 1;
        // stage current slice (registers -> LDS), fast: data already arrived
        {
            uint4* d0 = (uint4*)((unsigned int*)sWB[buf] + er0 * 20);
            uint4* d1 = (uint4*)((unsigned int*)sWB[buf] + er1 * 20);
            d0[ec0] = gc0;
            d1[ec1] = gc1;
        }
        // issue next slice's global loads (latency hidden by this iteration)
        if (idx + 1 < 36){
            const uint4* src = (const uint4*)(WB + (idx + 1) * 4096);
            gc0 = src[tid];
            gc1 = src[256 + tid];
        }

        const int t  = idx >> 2, kc = idx & 3;
        const int kh = t / 3,    kw = t % 3;
        const int ih = h + kh - 1;
        const bool do_comp = (ih >= 0) && (ih <= 127);   // uniform across block

        // B fragment: activations 32(K=c) x 16(N=pixel); lane lm = column,
        // lanes 0-15 K=0..15, lanes 16-31 K=16..31 (contiguous in NHWC).
        // Issue before the barrier so L2 latency overlaps store+sync.
        const int  iw    = wpix + kw - 1;
        const bool valid = (iw >= 0) && (iw < 128);
        FragU B;
        if (do_comp){
            const unsigned short* bbase =
                xmod + (size_t)((n * 128 + ih) * 128 + (valid ? iw : 0)) * 128;
            const uint4* bp = (const uint4*)(bbase + kc * 32 + hl * 16);
            uint4 b0 = bp[0], b1 = bp[1];
            if (!valid){ b0.x = b0.y = b0.z = b0.w = 0u;
                         b1.x = b1.y = b1.z = b1.w = 0u; }
            B.q[0] = b0; B.q[1] = b1;
        }

        __syncthreads();                   // staging of buf visible to all waves

        if (do_comp){
            // A fragments (weights 16(M=o) x 32(K=c)) for all 8 o-tiles:
            // issue the whole DS clause first, then run WMMAs back-to-back.
            FragU A[8];
            const unsigned short* sb = sWB[buf];
            #pragma unroll
            for (int ot = 0; ot < 8; ++ot){
                const uint4* ap = (const uint4*)(sb + (ot * 16 + lm) * 40 + hl * 16);
                A[ot].q[0] = ap[0];        // K = hl*16 + 0..7
                A[ot].q[1] = ap[1];        // K = hl*16 + 8..15
            }
            asm volatile("" ::: "memory"); // fence: keep DS loads ahead of WMMAs
            #pragma unroll
            for (int ot = 0; ot < 8; ++ot){
                acc[ot] = __builtin_amdgcn_wmma_f32_16x16x32_bf16(
                    false, A[ot].v, false, B.v, (short)0, acc[ot], false, false);
            }
        }
    }

    // Epilogue: D layout -> lane = pixel col, VGPR r = o-row (r + 8*hl)
    #pragma unroll
    for (int ot = 0; ot < 8; ++ot){
        int obase = ot * 16 + hl * 8;
        float4 s0 = *(const float4*)(s2t2 + obase);
        float4 s1 = *(const float4*)(s2t2 + obase + 4);
        float4 t0 = *(const float4*)(s2t2 + 128 + obase);
        float4 t1 = *(const float4*)(s2t2 + 128 + obase + 4);
        float sv[8] = {s0.x, s0.y, s0.z, s0.w, s1.x, s1.y, s1.z, s1.w};
        float tv[8] = {t0.x, t0.y, t0.z, t0.w, t1.x, t1.y, t1.z, t1.w};
        #pragma unroll
        for (int r = 0; r < 8; ++r){
            float v = acc[ot][r] * sv[r] + tv[r];
            v = v / (1.f + __expf(-v));     // SiLU
            out[(((size_t)n * 128 + (obase + r)) * 128 + h) * 128 + wpix] = v;
        }
    }
}

// ---------------- Launch ----------------
extern "C" void kernel_launch(void* const* d_in, const int* in_sizes, int n_in,
                              void* d_out, int out_size, void* d_ws, size_t ws_size,
                              hipStream_t stream) {
    (void)in_sizes; (void)n_in; (void)out_size; (void)ws_size;
    const float* x   = (const float*)d_in[0];
    const float* w1  = (const float*)d_in[1];
    const float* b1  = (const float*)d_in[2];
    const float* g1  = (const float*)d_in[3];
    const float* be1 = (const float*)d_in[4];
    const float* m1  = (const float*)d_in[5];
    const float* v1  = (const float*)d_in[6];
    const float* wh  = (const float*)d_in[7];
    const float* bh  = (const float*)d_in[8];
    const float* ww  = (const float*)d_in[9];
    const float* bw  = (const float*)d_in[10];
    const float* wc  = (const float*)d_in[11];
    const float* bc  = (const float*)d_in[12];
    const float* g2  = (const float*)d_in[13];
    const float* be2 = (const float*)d_in[14];
    const float* m2  = (const float*)d_in[15];
    const float* v2  = (const float*)d_in[16];
    float* out = (float*)d_out;

    char* ws = (char*)d_ws;
    float*          y_buf = (float*)ws;                                  // 1 MiB
    float*          a_buf = (float*)(ws + (1u  << 20));                  // 1 MiB
    unsigned short* xmod  = (unsigned short*)(ws + (2u  << 20));         // 32 MiB (bf16 NHWC)
    unsigned short* WB    = (unsigned short*)(ws + (34u << 20));         // 288 KiB
    float*          s2t2  = (float*)(ws + (34u << 20) + 294912u);        // 1 KiB

    pool_kernel<<<1024, 128, 0, stream>>>(x, y_buf);
    attn_kernel<<<8, 256, 0, stream>>>(y_buf, w1, b1, g1, be1, m1, v1,
                                       wh, bh, ww, bw, a_buf);
    prep_w_kernel<<<576, 256, 0, stream>>>(wc, WB);
    prep_s_kernel<<<1, 128, 0, stream>>>(bc, g2, be2, m2, v2, s2t2);
    mod_kernel<<<8 * 128 * 4, 256, 0, stream>>>(x, a_buf, (unsigned int*)xmod);
    conv_kernel<<<1024, 256, 0, stream>>>(xmod, WB, s2t2, out);
}
